// Net_89197880803784
// MI455X (gfx1250) — matrix operand order repaired and checked by reading, not compiled, not run on hardware
//
#include <hip/hip_runtime.h>
#include <math.h>

// ---- problem constants (from reference) ----
#define U_N    10000
#define T_N    3
#define DAYS_N 10
#define HID_N  128
#define C_N    200
#define EMI    100000
#define EMA    3200
#define OUT_N  3
#define MTILES 626      // ceil(10000/16) + 1

typedef __attribute__((ext_vector_type(16))) __bf16 v16bf;
typedef __attribute__((ext_vector_type(8)))  __bf16 v8bf;
typedef __attribute__((ext_vector_type(8)))  float  v8f;

__device__ inline v8f wmma_bf16(v16bf a, v16bf b, v8f c) {
  return __builtin_amdgcn_wmma_f32_16x16x32_bf16(false, a, false, b, (short)0, c,
                                                 false, false);
}

__device__ inline v16bf combine16(v8bf lo, v8bf hi) {
  return __builtin_shufflevector(lo, hi, 0, 1, 2, 3, 4, 5, 6, 7,
                                 8, 9, 10, 11, 12, 13, 14, 15);
}

// A fragment: elements 0..15 = row[base..base+15]
__device__ inline v16bf load_a_tile(const __bf16* row, int base) {
  v8bf lo = *(const v8bf*)(row + base);
  v8bf hi = *(const v8bf*)(row + base + 8);
  return combine16(lo, hi);
}
// B fragment: 16 contiguous bf16 at p (k-contiguous packed weights)
__device__ inline v16bf load_b_tile(const __bf16* p) {
  v8bf lo = *(const v8bf*)(p);
  v8bf hi = *(const v8bf*)(p + 8);
  return combine16(lo, hi);
}

__device__ inline float sigmf(float x) { return 1.f / (1.f + __expf(-x)); }

// ============================ utility / packing kernels ============================

__global__ void fill_f32(float* p, float v, long n) {
  long i = (long)blockIdx.x * blockDim.x + threadIdx.x;
  if (i < n) p[i] = v;
}

__global__ void cvt_bf_k(const float* s, __bf16* d, long n) {
  long i = (long)blockIdx.x * blockDim.x + threadIdx.x;
  if (i < n) d[i] = (__bf16)s[i];
}

// Wt[n][k] = attn_W[k][n], bf16, (128 x 384)
__global__ void pack_attnW_k(const float* W, __bf16* Wt) {
  int idx = blockIdx.x * blockDim.x + threadIdx.x;
  if (idx >= HID_N * 384) return;
  int n = idx / 384, k = idx % 384;
  Wt[idx] = (__bf16)W[(size_t)k * HID_N + n];
}

// kexp[m][j] = k_in[j%8][m], bf16, (U x 256)
__global__ void kexp_k(const float* kin, __bf16* kexp) {
  long idx = (long)blockIdx.x * blockDim.x + threadIdx.x;
  if (idx >= (long)U_N * 256) return;
  int m = (int)(idx / 256), j = (int)(idx % 256);
  kexp[idx] = (__bf16)kin[(size_t)(j & 7) * U_N + m];
}

__global__ void count_edges_k(const int* ei, float* deg) {
  int e = blockIdx.x * blockDim.x + threadIdx.x;
  if (e < EMI) atomicAdd(&deg[ei[EMI + e]], 1.f);
}

__global__ void macro_degw_k(const int* ce, const float* cw, float* degm) {
  int e = blockIdx.x * blockDim.x + threadIdx.x;
  if (e < EMA) atomicAdd(&degm[ce[EMA + e]], cw[e]);
}

__global__ void rsqrt_k(float* p, int n) {
  int i = blockIdx.x * blockDim.x + threadIdx.x;
  if (i < n) p[i] = rsqrtf(p[i]);
}

__global__ void count_com_k(const int* com, float* counts) {
  int u = blockIdx.x * blockDim.x + threadIdx.x;
  if (u < U_N) atomicAdd(&counts[com[u]], 1.f);
}

// ============================ micro GCN, factored (2 input features) ============
// xw[u] = f0(u)*W0 + f1(u)*W1  =>  scatter only the scalars a0, a1.

__global__ void micro_a0_edge_k(const int* ei, const float* md, const float* dinv,
                                float* a0, int t) {
  long idx = (long)blockIdx.x * blockDim.x + threadIdx.x;
  if (idx >= (long)DAYS_N * EMI) return;
  int day = (int)(idx / EMI), e = (int)(idx % EMI);
  int s = ei[e], d = ei[EMI + e];
  atomicAdd(&a0[(size_t)day * U_N + d],
            dinv[s] * dinv[d] * md[((size_t)s * T_N + t) * DAYS_N + day]);
}

__global__ void micro_a1_edge_k(const int* ei, const float* degs, const float* dinv,
                                float* a1) {
  int e = blockIdx.x * blockDim.x + threadIdx.x;
  if (e >= EMI) return;
  int s = ei[e], d = ei[EMI + e];
  atomicAdd(&a1[d], dinv[s] * dinv[d] * degs[s]);
}

__global__ void micro_a0_diag_k(const float* md, const float* dinv, float* a0, int t) {
  long idx = (long)blockIdx.x * blockDim.x + threadIdx.x;
  if (idx >= (long)DAYS_N * U_N) return;
  int day = (int)(idx / U_N), u = (int)(idx % U_N);
  a0[idx] += dinv[u] * dinv[u] * md[((size_t)u * T_N + t) * DAYS_N + day];
}

__global__ void micro_a1_diag_k(const float* degs, const float* dinv, float* a1) {
  int u = blockIdx.x * blockDim.x + threadIdx.x;
  if (u < U_N) a1[u] += dinv[u] * dinv[u] * degs[u];
}

// community sums of the factored scalars (for Z)
__global__ void za0_k(const float* a0, const int* com, float* za0) {
  long idx = (long)blockIdx.x * blockDim.x + threadIdx.x;
  if (idx >= (long)DAYS_N * U_N) return;
  int day = (int)(idx / U_N), u = (int)(idx % U_N);
  atomicAdd(&za0[(size_t)day * C_N + com[u]], a0[idx]);
}

__global__ void za1_k(const float* a1, const int* com, float* za1) {
  int u = blockIdx.x * blockDim.x + threadIdx.x;
  if (u < U_N) atomicAdd(&za1[com[u]], a1[u]);
}

// Hr = relu(a0*W0 + a1*W1 + b) -> bf16, directly
__global__ void build_H_k(const float* a0, const float* a1, const float* gw,
                          const float* gb, __bf16* Hbf) {
  long idx = (long)blockIdx.x * blockDim.x + threadIdx.x;
  if (idx >= (long)DAYS_N * U_N * HID_N) return;
  int day = (int)(idx / ((long)U_N * HID_N));
  int u = (int)((idx / HID_N) % U_N);
  int h = (int)(idx % HID_N);
  float v = a0[(size_t)day * U_N + u] * gw[h] + a1[u] * gw[HID_N + h] + gb[h];
  Hbf[idx] = (__bf16)fmaxf(v, 0.f);
}

// Z = (za0/cnt)*W0 + (za1/cnt)*W1 + b ; fp32 (for Q) + bf16 (for attn)
__global__ void build_Z_k(const float* za0, const float* za1, const float* counts,
                          const float* gw, const float* gb, float* Zf, __bf16* Zbf) {
  long idx = (long)blockIdx.x * blockDim.x + threadIdx.x;
  if (idx >= (long)DAYS_N * C_N * HID_N) return;
  int day = (int)(idx / ((long)C_N * HID_N));
  int c = (int)((idx / HID_N) % C_N);
  int h = (int)(idx % HID_N);
  float inv = 1.f / counts[c];
  float z = (za0[(size_t)day * C_N + c] * inv) * gw[h]
          + (za1[c] * inv) * gw[HID_N + h] + gb[h];
  Zf[idx] = z;
  Zbf[idx] = (__bf16)z;
}

// ============================ macro GCN, factored (3 input features) ============

__global__ void macro_b01_edge_k(const int* ce, const float* cw, const float* dinvM,
                                 const float* mdyn, float* b0, float* b1, int t) {
  long idx = (long)blockIdx.x * blockDim.x + threadIdx.x;
  if (idx >= (long)DAYS_N * EMA) return;
  int day = (int)(idx / EMA), e = (int)(idx % EMA);
  int s = ce[e], d = ce[EMA + e];
  float coef = dinvM[s] * dinvM[d] * cw[e];
  const float* m = mdyn + (((size_t)s * T_N + t) * DAYS_N + day) * 2;
  atomicAdd(&b0[(size_t)day * C_N + d], coef * m[0]);
  atomicAdd(&b1[(size_t)day * C_N + d], coef * m[1]);
}

__global__ void macro_b2_edge_k(const int* ce, const float* cw, const float* dinvM,
                                const float* mstat, float* b2) {
  int e = blockIdx.x * blockDim.x + threadIdx.x;
  if (e >= EMA) return;
  int s = ce[e], d = ce[EMA + e];
  atomicAdd(&b2[d], dinvM[s] * dinvM[d] * cw[e] * mstat[s]);
}

__global__ void macro_b01_diag_k(const float* mdyn, const float* dinvM,
                                 float* b0, float* b1, int t) {
  long idx = (long)blockIdx.x * blockDim.x + threadIdx.x;
  if (idx >= (long)DAYS_N * C_N) return;
  int day = (int)(idx / C_N), c = (int)(idx % C_N);
  float di2 = dinvM[c] * dinvM[c];
  const float* m = mdyn + (((size_t)c * T_N + t) * DAYS_N + day) * 2;
  b0[idx] += di2 * m[0];
  b1[idx] += di2 * m[1];
}

__global__ void macro_b2_diag_k(const float* mstat, const float* dinvM, float* b2) {
  int c = blockIdx.x * blockDim.x + threadIdx.x;
  if (c < C_N) b2[c] += dinvM[c] * dinvM[c] * mstat[c];
}

__global__ void build_Y_k(const float* b0, const float* b1, const float* b2,
                          const float* mw, const float* mb, float* Yf, __bf16* Ybf) {
  long idx = (long)blockIdx.x * blockDim.x + threadIdx.x;
  if (idx >= (long)DAYS_N * C_N * HID_N) return;
  int day = (int)(idx / ((long)C_N * HID_N));
  int c = (int)((idx / HID_N) % C_N);
  int h = (int)(idx % HID_N);
  long dc = (long)day * C_N + c;
  float y = b0[dc] * mw[h] + b1[dc] * mw[HID_N + h] + b2[c] * mw[2 * HID_N + h] + mb[h];
  Yf[idx] = y;
  Ybf[idx] = (__bf16)y;
}

__global__ void q_k(const float* Z, const float* Yb, float* Q) {
  long idx = (long)blockIdx.x * blockDim.x + threadIdx.x;
  if (idx >= (long)DAYS_N * C_N * 256) return;
  long dc = idx / 256;
  int j = (int)(idx % 256);
  float v = (j < HID_N) ? Z[dc * HID_N + j] : Yb[dc * HID_N + (j - HID_N)];
  Q[idx] = fmaxf(v, 0.f);
}

// ============================ WMMA attention GEMM ============================
// out = relu([A | S1row | S2row] @ W + b) * Mult ; K = 384, all-bf16 b128 loads.
// Row of S1/S2 selected by com[m] (comp!=null) or m itself (comp==null).
__global__ __launch_bounds__(32) void attn_k(
    const __bf16* __restrict__ Aseq,
    const __bf16* __restrict__ S1, long s1_day, int s1_row,
    const __bf16* __restrict__ S2, long s2_day, int s2_row,
    const int* __restrict__ comp,
    const __bf16* __restrict__ Wt,     // (128 x 384) bf16, k-contiguous rows
    const float* __restrict__ bias, const __bf16* __restrict__ Mult,
    __bf16* __restrict__ Outp) {
  const int lane = threadIdx.x;
  const int tm = blockIdx.x, tn = blockIdx.y, day = blockIdx.z;
  int mrow = tm * 16 + (lane & 15);
  if (mrow >= U_N) mrow = U_N - 1;          // clamp: invalid rows never stored
  const int half8 = ((lane >> 4) & 1) * 16;
  const int n = tn * 16 + (lane & 15);
  const int kb = ((lane >> 4) & 1) * 16;

  const __bf16* r0 = Aseq + ((size_t)day * U_N + mrow) * HID_N;
  const int ridx = comp ? comp[mrow] : mrow;
  const __bf16* r1 = S1 + (size_t)day * s1_day + (size_t)ridx * s1_row;
  const __bf16* r2 = S2 + (size_t)day * s2_day + (size_t)ridx * s2_row;
  const __bf16* wr = Wt + (size_t)n * 384 + kb;

  v8f acc = {0.f, 0.f, 0.f, 0.f, 0.f, 0.f, 0.f, 0.f};
#pragma unroll
  for (int kk = 0; kk < HID_N; kk += 32)
    acc = wmma_bf16(load_a_tile(r0, kk + half8), load_b_tile(wr + kk), acc);
#pragma unroll
  for (int kk = 0; kk < HID_N; kk += 32)
    acc = wmma_bf16(load_a_tile(r1, kk + half8), load_b_tile(wr + 128 + kk), acc);
#pragma unroll
  for (int kk = 0; kk < HID_N; kk += 32)
    acc = wmma_bf16(load_a_tile(r2, kk + half8), load_b_tile(wr + 256 + kk), acc);

  const int half = (lane >> 4) & 1;
  const float bn = bias[n];
#pragma unroll
  for (int r = 0; r < 8; ++r) {
    int m = tm * 16 + r + half * 8;
    if (m < U_N) {
      float v = acc[r] + bn;
      v = v > 0.f ? v : 0.f;
      size_t off = ((size_t)day * U_N + m) * HID_N + n;
      Outp[off] = (__bf16)(v * (float)Mult[off]);
    }
  }
}

// ============================ WMMA LSTM gate GEMM ============================
__global__ __launch_bounds__(32) void lstm_gemm_k(
    const __bf16* __restrict__ X, const __bf16* __restrict__ Hprev,
    const __bf16* __restrict__ Wihb, const __bf16* __restrict__ Whhb, // (512x128) bf16
    const float* __restrict__ bih, const float* __restrict__ bhh,
    float* __restrict__ gates) {
  const int lane = threadIdx.x;
  const int tm = blockIdx.x, tn = blockIdx.y;
  int mrow = tm * 16 + (lane & 15);
  if (mrow >= U_N) mrow = U_N - 1;
  const int half8 = ((lane >> 4) & 1) * 16;
  const int n = tn * 16 + (lane & 15);
  const int kb = ((lane >> 4) & 1) * 16;

  v8f acc = {0.f, 0.f, 0.f, 0.f, 0.f, 0.f, 0.f, 0.f};
  const __bf16* xr = X + (size_t)mrow * HID_N;
  const __bf16* w1 = Wihb + (size_t)n * HID_N + kb;
#pragma unroll
  for (int kk = 0; kk < HID_N; kk += 32)
    acc = wmma_bf16(load_a_tile(xr, kk + half8), load_b_tile(w1 + kk), acc);
  if (Hprev != (const __bf16*)0) {
    const __bf16* hr = Hprev + (size_t)mrow * HID_N;
    const __bf16* w2 = Whhb + (size_t)n * HID_N + kb;
#pragma unroll
    for (int kk = 0; kk < HID_N; kk += 32)
      acc = wmma_bf16(load_a_tile(hr, kk + half8), load_b_tile(w2 + kk), acc);
  }
  const float bn = bih[n] + bhh[n];
  const int half = (lane >> 4) & 1;
#pragma unroll
  for (int r = 0; r < 8; ++r) {
    int m = tm * 16 + r + half * 8;
    if (m < U_N) gates[(size_t)m * 512 + n] = acc[r] + bn;
  }
}

__global__ void lstm_cell_k(const float* gates, float* cbuf, __bf16* hout) {
  long idx = (long)blockIdx.x * blockDim.x + threadIdx.x;
  if (idx >= (long)U_N * HID_N) return;
  int u = (int)(idx / HID_N), h = (int)(idx % HID_N);
  const float* g = gates + (size_t)u * 512;
  float ig = sigmf(g[h]);
  float fg = sigmf(g[HID_N + h]);
  float gg = tanhf(g[2 * HID_N + h]);
  float og = sigmf(g[3 * HID_N + h]);
  float c = fg * cbuf[idx] + ig * gg;
  cbuf[idx] = c;
  hout[idx] = (__bf16)(og * tanhf(c));
}

// ============================ macro LSTM (hidden=2) ============================
__global__ void macro_lstm_k(const float* Q, const float* Wih0, const float* Whh0,
                             const float* bih0, const float* bhh0,
                             const float* Wih1, const float* Whh1,
                             const float* bih1, const float* bhh1, float* out) {
  int c = blockIdx.x * blockDim.x + threadIdx.x;
  if (c >= C_N) return;
  float y0[DAYS_N][2];
  float h0 = 0.f, h1 = 0.f, c0 = 0.f, c1 = 0.f;
  for (int d = 0; d < DAYS_N; ++d) {
    const float* q = Q + ((size_t)d * C_N + c) * 256;
    float g[8];
    for (int r = 0; r < 8; ++r) {
      float s = bih0[r] + bhh0[r];
      for (int k = 0; k < 256; ++k) s += q[k] * Wih0[r * 256 + k];
      s += h0 * Whh0[r * 2 + 0] + h1 * Whh0[r * 2 + 1];
      g[r] = s;
    }
    c0 = sigmf(g[2]) * c0 + sigmf(g[0]) * tanhf(g[4]);
    c1 = sigmf(g[3]) * c1 + sigmf(g[1]) * tanhf(g[5]);
    h0 = sigmf(g[6]) * tanhf(c0);
    h1 = sigmf(g[7]) * tanhf(c1);
    y0[d][0] = h0; y0[d][1] = h1;
  }
  h0 = h1 = c0 = c1 = 0.f;
  for (int d = 0; d < DAYS_N; ++d) {
    float g[8];
    for (int r = 0; r < 8; ++r)
      g[r] = bih1[r] + bhh1[r] + y0[d][0] * Wih1[r * 2 + 0] + y0[d][1] * Wih1[r * 2 + 1]
           + h0 * Whh1[r * 2 + 0] + h1 * Whh1[r * 2 + 1];
    c0 = sigmf(g[2]) * c0 + sigmf(g[0]) * tanhf(g[4]);
    c1 = sigmf(g[3]) * c1 + sigmf(g[1]) * tanhf(g[5]);
    h0 = sigmf(g[6]) * tanhf(c0);
    h1 = sigmf(g[7]) * tanhf(c1);
  }
  out[c * 2 + 0] = h0;
  out[c * 2 + 1] = h1;
}

// ============================ final GCN + softmax ============================
__global__ void final_xw_k(const __bf16* Htk, const float* Wout, float* xw3) {
  int u = blockIdx.x * blockDim.x + threadIdx.x;
  if (u >= U_N) return;
  const __bf16* row = Htk + (size_t)u * (DAYS_N * HID_N);
  float s0 = 0.f, s1 = 0.f, s2 = 0.f;
  for (int j = 0; j < DAYS_N * HID_N; j += 8) {
    v8bf v = *(const v8bf*)(row + j);
#pragma unroll
    for (int i = 0; i < 8; ++i) {
      float x = fmaxf((float)v[i], 0.f);
      const float* w = Wout + (size_t)(j + i) * OUT_N;
      s0 += x * w[0]; s1 += x * w[1]; s2 += x * w[2];
    }
  }
  xw3[u * OUT_N + 0] = s0;
  xw3[u * OUT_N + 1] = s1;
  xw3[u * OUT_N + 2] = s2;
}

__global__ void final_edge_k(const int* ei, const float* dinv, const float* xw3,
                             float* accOut) {
  long idx = (long)blockIdx.x * blockDim.x + threadIdx.x;
  if (idx >= (long)EMI * OUT_N) return;
  int e = (int)(idx / OUT_N), o = (int)(idx % OUT_N);
  int s = ei[e], d = ei[EMI + e];
  atomicAdd(&accOut[d * OUT_N + o], dinv[s] * dinv[d] * xw3[s * OUT_N + o]);
}

__global__ void final_out_k(const float* accOut, const float* xw3, const float* dinv,
                            const float* bout, float* outp) {
  int u = blockIdx.x * blockDim.x + threadIdx.x;
  if (u >= U_N) return;
  float di2 = dinv[u] * dinv[u];
  float v[OUT_N];
  float mx = -1e30f;
  for (int o = 0; o < OUT_N; ++o) {
    v[o] = accOut[u * OUT_N + o] + di2 * xw3[u * OUT_N + o] + bout[o];
    mx = fmaxf(mx, v[o]);
  }
  float sum = 0.f;
  for (int o = 0; o < OUT_N; ++o) { v[o] = __expf(v[o] - mx); sum += v[o]; }
  for (int o = 0; o < OUT_N; ++o) outp[u * OUT_N + o] = v[o] / sum;
}

// ============================ host side ============================

static inline dim3 g1(long n) { return dim3((unsigned)((n + 255) / 256)); }

extern "C" void kernel_launch(void* const* d_in, const int* in_sizes, int n_in,
                              void* d_out, int out_size, void* d_ws, size_t ws_size,
                              hipStream_t stream) {
  const float* md    = (const float*)d_in[0];
  const int*   ei    = (const int*)d_in[1];
  const float* degs  = (const float*)d_in[2];
  const int*   com   = (const int*)d_in[3];
  const int*   ce    = (const int*)d_in[4];
  const float* cw    = (const float*)d_in[5];
  const float* mdyn  = (const float*)d_in[6];
  const float* mstat = (const float*)d_in[7];
  const float* kin   = (const float*)d_in[8];
  const float* gcn_W  = (const float*)d_in[9];
  const float* gcn_b  = (const float*)d_in[10];
  const float* mgcn_W = (const float*)d_in[11];
  const float* mgcn_b = (const float*)d_in[12];
  const float* attn_W = (const float*)d_in[13];
  const float* attn_b = (const float*)d_in[14];
  const float* out_W  = (const float*)d_in[15];
  const float* out_b  = (const float*)d_in[16];
  const float* l_Wih[2] = {(const float*)d_in[17], (const float*)d_in[21]};
  const float* l_Whh[2] = {(const float*)d_in[18], (const float*)d_in[22]};
  const float* l_bih[2] = {(const float*)d_in[19], (const float*)d_in[23]};
  const float* l_bhh[2] = {(const float*)d_in[20], (const float*)d_in[24]};
  const float* m0_Wih = (const float*)d_in[25];
  const float* m0_Whh = (const float*)d_in[26];
  const float* m0_bih = (const float*)d_in[27];
  const float* m0_bhh = (const float*)d_in[28];
  const float* m1_Wih = (const float*)d_in[29];
  const float* m1_Whh = (const float*)d_in[30];
  const float* m1_bih = (const float*)d_in[31];
  const float* m1_bhh = (const float*)d_in[32];
  float* outp = (float*)d_out;

  // ---- workspace carve-up ----
  char* ws = (char*)d_ws;
  size_t off = 0;
  auto carve = [&](size_t bytes) -> void* {
    void* p = ws + off;
    off = (off + bytes + 255) & ~(size_t)255;
    return p;
  };
  const long NUH = (long)DAYS_N * U_N * HID_N;   // 12.8M
  const long NCH = (long)DAYS_N * C_N * HID_N;   // 256K
  __bf16* Hbf  = (__bf16*)carve((size_t)NUH * 2);      // 25.6 MB
  __bf16* Hv1  = (__bf16*)carve((size_t)NUH * 2);
  __bf16* Hv2  = (__bf16*)carve((size_t)NUH * 2);
  float*  gates = (float*)carve((size_t)U_N * 512 * 4);      // 20.5 MB
  float*  cbuf  = (float*)carve((size_t)U_N * HID_N * 4);    // 5.1 MB
  __bf16* kexp  = (__bf16*)carve((size_t)U_N * 256 * 2);     // 5.1 MB
  __bf16* Wt    = (__bf16*)carve((size_t)HID_N * 384 * 2);
  __bf16* Wihb[2] = {(__bf16*)carve((size_t)512 * HID_N * 2),
                     (__bf16*)carve((size_t)512 * HID_N * 2)};
  __bf16* Whhb[2] = {(__bf16*)carve((size_t)512 * HID_N * 2),
                     (__bf16*)carve((size_t)512 * HID_N * 2)};
  float*  Zf  = (float*)carve((size_t)NCH * 4);
  float*  Yf  = (float*)carve((size_t)NCH * 4);
  __bf16* Zbf = (__bf16*)carve((size_t)NCH * 2);
  __bf16* Ybf = (__bf16*)carve((size_t)NCH * 2);
  float*  Qb  = (float*)carve((size_t)DAYS_N * C_N * 256 * 4);
  float*  a0  = (float*)carve((size_t)DAYS_N * U_N * 4);
  float*  a1  = (float*)carve((size_t)U_N * 4);
  float*  za0 = (float*)carve((size_t)DAYS_N * C_N * 4);
  float*  za1 = (float*)carve((size_t)C_N * 4);
  float*  b0  = (float*)carve((size_t)DAYS_N * C_N * 4);
  float*  b1  = (float*)carve((size_t)DAYS_N * C_N * 4);
  float*  b2  = (float*)carve((size_t)C_N * 4);
  float*  dinv   = (float*)carve((size_t)U_N * 4);
  float*  counts = (float*)carve((size_t)C_N * 4);
  float*  dinvM  = (float*)carve((size_t)C_N * 4);
  float*  xw3    = (float*)carve((size_t)U_N * OUT_N * 4);
  float*  accOut = (float*)carve((size_t)U_N * OUT_N * 4);

  // ---- one-time precompute ----
  fill_f32<<<g1(U_N), 256, 0, stream>>>(dinv, 1.f, U_N);
  count_edges_k<<<g1(EMI), 256, 0, stream>>>(ei, dinv);
  rsqrt_k<<<g1(U_N), 256, 0, stream>>>(dinv, U_N);
  fill_f32<<<g1(C_N), 256, 0, stream>>>(counts, 0.f, C_N);
  count_com_k<<<g1(U_N), 256, 0, stream>>>(com, counts);
  fill_f32<<<g1(C_N), 256, 0, stream>>>(dinvM, 1.f, C_N);
  macro_degw_k<<<g1(EMA), 256, 0, stream>>>(ce, cw, dinvM);
  rsqrt_k<<<g1(C_N), 256, 0, stream>>>(dinvM, C_N);

  pack_attnW_k<<<g1((long)HID_N * 384), 256, 0, stream>>>(attn_W, Wt);
  kexp_k<<<g1((long)U_N * 256), 256, 0, stream>>>(kin, kexp);
  for (int l = 0; l < 2; ++l) {
    cvt_bf_k<<<g1((long)512 * HID_N), 256, 0, stream>>>(l_Wih[l], Wihb[l], (long)512 * HID_N);
    cvt_bf_k<<<g1((long)512 * HID_N), 256, 0, stream>>>(l_Whh[l], Whhb[l], (long)512 * HID_N);
  }

  // t-independent factored scalars
  fill_f32<<<g1(U_N), 256, 0, stream>>>(a1, 0.f, U_N);
  micro_a1_edge_k<<<g1(EMI), 256, 0, stream>>>(ei, degs, dinv, a1);
  micro_a1_diag_k<<<g1(U_N), 256, 0, stream>>>(degs, dinv, a1);
  fill_f32<<<g1(C_N), 256, 0, stream>>>(za1, 0.f, C_N);
  za1_k<<<g1(U_N), 256, 0, stream>>>(a1, com, za1);
  fill_f32<<<g1(C_N), 256, 0, stream>>>(b2, 0.f, C_N);
  macro_b2_edge_k<<<g1(EMA), 256, 0, stream>>>(ce, cw, dinvM, mstat, b2);
  macro_b2_diag_k<<<g1(C_N), 256, 0, stream>>>(mstat, dinvM, b2);

  for (int t = 0; t < T_N; ++t) {
    // micro GCN (factored)
    fill_f32<<<g1((long)DAYS_N * U_N), 256, 0, stream>>>(a0, 0.f, (long)DAYS_N * U_N);
    micro_a0_edge_k<<<g1((long)DAYS_N * EMI), 256, 0, stream>>>(ei, md, dinv, a0, t);
    micro_a0_diag_k<<<g1((long)DAYS_N * U_N), 256, 0, stream>>>(md, dinv, a0, t);
    fill_f32<<<g1((long)DAYS_N * C_N), 256, 0, stream>>>(za0, 0.f, (long)DAYS_N * C_N);
    za0_k<<<g1((long)DAYS_N * U_N), 256, 0, stream>>>(a0, com, za0);
    build_H_k<<<g1(NUH), 256, 0, stream>>>(a0, a1, gcn_W, gcn_b, Hbf);
    build_Z_k<<<g1(NCH), 256, 0, stream>>>(za0, za1, counts, gcn_W, gcn_b, Zf, Zbf);
    // macro GCN (factored)
    fill_f32<<<g1((long)DAYS_N * C_N), 256, 0, stream>>>(b0, 0.f, (long)DAYS_N * C_N);
    fill_f32<<<g1((long)DAYS_N * C_N), 256, 0, stream>>>(b1, 0.f, (long)DAYS_N * C_N);
    macro_b01_edge_k<<<g1((long)DAYS_N * EMA), 256, 0, stream>>>(ce, cw, dinvM, mdyn, b0, b1, t);
    macro_b01_diag_k<<<g1((long)DAYS_N * C_N), 256, 0, stream>>>(mdyn, dinvM, b0, b1, t);
    build_Y_k<<<g1(NCH), 256, 0, stream>>>(b0, b1, b2, mgcn_W, mgcn_b, Yf, Ybf);
    q_k<<<g1((long)DAYS_N * C_N * 256), 256, 0, stream>>>(Zf, Yf, Qb);

    // two WMMA attentions (all-bf16, b128 tile loads)
    dim3 ag(MTILES, HID_N / 16, DAYS_N);
    attn_k<<<ag, 32, 0, stream>>>(Hbf,
                                  Zbf, (long)C_N * HID_N, HID_N,
                                  Ybf, (long)C_N * HID_N, HID_N,
                                  com, Wt, attn_b, Hbf, Hv1);
    attn_k<<<ag, 32, 0, stream>>>(Hv1,
                                  kexp, 0L, 256,
                                  kexp + HID_N, 0L, 256,
                                  (const int*)0, Wt, attn_b, Hv1, Hv2);

    // 2-layer LSTM over days: layer0 Hv2->Hv1, layer1 Hv1->Hv2
    __bf16* seq_in = Hv2;
    __bf16* seq_out = Hv1;
    for (int layer = 0; layer < 2; ++layer) {
      fill_f32<<<g1((long)U_N * HID_N), 256, 0, stream>>>(cbuf, 0.f, (long)U_N * HID_N);
      for (int d = 0; d < DAYS_N; ++d) {
        const __bf16* hprev = d ? (seq_out + (size_t)(d - 1) * U_N * HID_N) : (const __bf16*)0;
        lstm_gemm_k<<<dim3(MTILES, 32), 32, 0, stream>>>(
            seq_in + (size_t)d * U_N * HID_N, hprev,
            Wihb[layer], Whhb[layer], l_bih[layer], l_bhh[layer], gates);
        lstm_cell_k<<<g1((long)U_N * HID_N), 256, 0, stream>>>(
            gates, cbuf, seq_out + (size_t)d * U_N * HID_N);
      }
      __bf16* tmp = seq_in; seq_in = seq_out; seq_out = tmp;
    }
    // Htk now in Hv2

    macro_lstm_k<<<g1(C_N), 256, 0, stream>>>(Qb, m0_Wih, m0_Whh, m0_bih, m0_bhh,
                                              m1_Wih, m1_Whh, m1_bih, m1_bhh,
                                              outp + (size_t)T_N * U_N * OUT_N + (size_t)t * C_N * 2);

    final_xw_k<<<g1(U_N), 256, 0, stream>>>(Hv2, out_W, xw3);
    fill_f32<<<g1((long)U_N * OUT_N), 256, 0, stream>>>(accOut, 0.f, (long)U_N * OUT_N);
    final_edge_k<<<g1((long)EMI * OUT_N), 256, 0, stream>>>(ei, dinv, xw3, accOut);
    final_out_k<<<g1(U_N), 256, 0, stream>>>(accOut, xw3, dinv, out_b,
                                             outp + (size_t)t * U_N * OUT_N);
  }
  (void)in_sizes; (void)n_in; (void)out_size; (void)ws_size;
}
